// GNNWithRegularization_52699248722264
// MI455X (gfx1250) — matrix operand order repaired and checked by reading, not compiled, not run on hardware
//
#include <hip/hip_runtime.h>
#include <hip/hip_bf16.h>

// ---------------------------------------------------------------------------
// Types for gfx1250 WMMA (wave32). 16x16x32 bf16 -> f32 accumulate.
// ---------------------------------------------------------------------------
typedef __attribute__((ext_vector_type(16))) __bf16        bf16x16;
typedef __attribute__((ext_vector_type(8)))  float         floatx8;
typedef __attribute__((ext_vector_type(8)))  unsigned int  uintx8;
typedef __attribute__((ext_vector_type(4)))  int           intx4;

#define D_IN  128
#define D_H   128
#define D_OUT 64
#define BN_EPS 1e-5f

static __device__ __forceinline__ unsigned short f2bf_bits(float f) {
    unsigned int u = __float_as_uint(f);
    unsigned int r = (u + 0x7FFFu + ((u >> 16) & 1u)) >> 16;  // RNE truncate
    return (unsigned short)r;
}

// ---------------------------------------------------------------------------
// Async global->LDS 16B copy (CDNA5 GLOBAL_LOAD_ASYNC_TO_LDS_B128, ASYNCcnt),
// with a register-copy fallback if the builtin is absent.
// ---------------------------------------------------------------------------
static __device__ __forceinline__ void async_copy16(const unsigned short* g,
                                                    unsigned short* l) {
#if __has_builtin(__builtin_amdgcn_global_load_async_to_lds_b128)
    __builtin_amdgcn_global_load_async_to_lds_b128(
        (__attribute__((address_space(1))) intx4*)g,
        (__attribute__((address_space(3))) intx4*)l, 0, 0);
#else
    *(uint4*)l = *(const uint4*)g;
#endif
}

static __device__ __forceinline__ void wait_async_copies() {
#if __has_builtin(__builtin_amdgcn_global_load_async_to_lds_b128)
#if __has_builtin(__builtin_amdgcn_s_wait_asynccnt)
    __builtin_amdgcn_s_wait_asynccnt(0);
#else
    asm volatile("s_wait_asynccnt 0x0" ::: "memory");
#endif
#endif
}

// A fragment (16x32 bf16) from LDS row-major tile (stride `ld` ushorts).
// Per ISA 7.12.2, lane = g*16+m holds row m; halves 0-7 <-> K = k0+g*8+0..7,
// halves 8-15 <-> K = k0+16+g*8+0..7 : two contiguous 16B runs -> 2x b128.
static __device__ __forceinline__ bf16x16
load_a_frag_lds(const unsigned short* As, int ld, int r0, int k0, int lane) {
    const int m = lane & 15;
    const int g = lane >> 4;
    const unsigned short* row = As + (r0 + m) * ld + k0 + (g << 3);
    union { uint4 q[2]; bf16x16 v; } u;
    u.q[0] = *(const uint4*)(row);
    u.q[1] = *(const uint4*)(row + 16);
    return u.v;
}

// B fragment from fragment-packed weights: frag f = (k0/32)*CT + (c0/16),
// lane's 16 halves contiguous (32B) -> 2x global_load_b128, zero repacking.
static __device__ __forceinline__ bf16x16
load_b_frag_packed(const unsigned short* Wp, int f, int lane) {
    union { uintx8 q; bf16x16 v; } u;
    u.q = *(const uintx8*)(Wp + (((long long)f * 32 + lane) << 4));
    return u.v;
}

// ---------------------------------------------------------------------------
// Utility kernels
// ---------------------------------------------------------------------------
__global__ void zero_f32_kernel(float* __restrict__ p, long long n) {
    long long i = (long long)blockIdx.x * blockDim.x + threadIdx.x;
    long long stride = (long long)gridDim.x * blockDim.x;
    for (; i < n; i += stride) p[i] = 0.0f;
}

// Pack W[128 x COLS] (fp32, K-major) into bf16 WMMA-B fragment order.
// dst[((kt*CT + ct)*32 + l)*16 + h] = bf16(W[(kt*32 + (h>>3)*16 + (l>>4)*8 + (h&7))*COLS + ct*16 + (l&15)])
__global__ void pack_w_kernel(const float* __restrict__ W,
                              unsigned short* __restrict__ dst,
                              int COLS, int total) {
    int i = blockIdx.x * blockDim.x + threadIdx.x;
    if (i >= total) return;
    const int CT = COLS >> 4;
    int h = i & 15;
    int l = (i >> 4) & 31;
    int f = i >> 9;
    int ct = f % CT;
    int kt = f / CT;
    int k = (kt << 5) + ((h >> 3) << 4) + ((l >> 4) << 3) + (h & 7);
    int c = (ct << 4) + (l & 15);
    dst[i] = f2bf_bits(W[k * COLS + c]);
}

// Xbf[row][c] = bf16(X[row][c] * rowscale[row]); zero for padded rows >= N.
__global__ void scale_to_bf16_kernel(const float* __restrict__ X,
                                     const float* __restrict__ rowscale,
                                     unsigned short* __restrict__ Xbf,
                                     int N, long long total) {
    long long i = (long long)blockIdx.x * blockDim.x + threadIdx.x;
    long long stride = (long long)gridDim.x * blockDim.x;
    for (; i < total; i += stride) {
        int r = (int)(i >> 7);
        float v = 0.0f;
        if (r < N) v = X[i] * rowscale[r];
        Xbf[i] = f2bf_bits(v);
    }
}

// degrees of the masked subgraph
__global__ void degree_kernel(const int* __restrict__ src,
                              const int* __restrict__ dst,
                              const unsigned char* __restrict__ mask,
                              float* __restrict__ out_deg,
                              float* __restrict__ in_deg, int E) {
    int e = blockIdx.x * blockDim.x + threadIdx.x;
    if (e >= E) return;
    if (!mask[e]) return;
    __hip_atomic_fetch_add(&out_deg[src[e]], 1.0f, __ATOMIC_RELAXED,
                           __HIP_MEMORY_SCOPE_AGENT);
    __hip_atomic_fetch_add(&in_deg[dst[e]], 1.0f, __ATOMIC_RELAXED,
                           __HIP_MEMORY_SCOPE_AGENT);
}

// in-place deg -> rsqrt(max(deg,1))
__global__ void norm_kernel(float* __restrict__ p, int n) {
    int i = blockIdx.x * blockDim.x + threadIdx.x;
    if (i < n) p[i] = __frsqrt_rn(fmaxf(p[i], 1.0f));
}

// ---------------------------------------------------------------------------
// WMMA GEMM: Y[N x COLS] = Xbf[padded N x 128] @ Wp (fragment-packed bf16).
// Block: 256 threads (8 waves), 128 rows per block, 16 rows per wave.
// A tile: async global->LDS b128 copies; B: packed 32B loads from L2-hot W.
// ---------------------------------------------------------------------------
template <int COLS>
__global__ __launch_bounds__(256) void
gemm_wmma_kernel(const unsigned short* __restrict__ Xbf,
                 const unsigned short* __restrict__ Wp,
                 float* __restrict__ Y, int N) {
    constexpr int LDA = D_IN + 8;  // 136 ushorts = 272B, 16B-aligned rows
    constexpr int CT = COLS >> 4;
    __shared__ __align__(32) unsigned short As[128 * LDA];

    const int tid = threadIdx.x;
    const int row0 = blockIdx.x * 128;

    // stage A tile: 128 rows x 256B as 16B chunks (2048 chunks, 8 per thread)
#pragma unroll
    for (int q = tid; q < 128 * 16; q += 256) {
        int r = q >> 4;        // tile row
        int c16 = q & 15;      // 16B chunk within row
        const unsigned short* g =
            Xbf + ((long long)(row0 + r) << 7) + (c16 << 3);
        unsigned short* l = As + r * LDA + (c16 << 3);
        async_copy16(g, l);
    }
    wait_async_copies();
    __syncthreads();

    const int wave = tid >> 5;
    const int lane = tid & 31;
    const int r0 = wave * 16;

    // hoist the 4 K-step A fragments for this wave's 16 rows
    bf16x16 a0 = load_a_frag_lds(As, LDA, r0, 0,  lane);
    bf16x16 a1 = load_a_frag_lds(As, LDA, r0, 32, lane);
    bf16x16 a2 = load_a_frag_lds(As, LDA, r0, 64, lane);
    bf16x16 a3 = load_a_frag_lds(As, LDA, r0, 96, lane);

    const int n = lane & 15;
    const int g = lane >> 4;
    const bool full = (row0 + 128) <= N;

#pragma unroll
    for (int ct = 0; ct < CT; ++ct) {
        const int c0 = ct << 4;
        floatx8 acc = {};
        acc = __builtin_amdgcn_wmma_f32_16x16x32_bf16(
            false, a0, false, load_b_frag_packed(Wp, 0 * CT + ct, lane),
            (short)0, acc, false, false);
        acc = __builtin_amdgcn_wmma_f32_16x16x32_bf16(
            false, a1, false, load_b_frag_packed(Wp, 1 * CT + ct, lane),
            (short)0, acc, false, false);
        acc = __builtin_amdgcn_wmma_f32_16x16x32_bf16(
            false, a2, false, load_b_frag_packed(Wp, 2 * CT + ct, lane),
            (short)0, acc, false, false);
        acc = __builtin_amdgcn_wmma_f32_16x16x32_bf16(
            false, a3, false, load_b_frag_packed(Wp, 3 * CT + ct, lane),
            (short)0, acc, false, false);

        // C/D layout: float v -> row = r0 + v + 8*g, col = c0 + n
        float* yb = Y + (long long)(row0 + r0 + (g << 3)) * COLS + c0 + n;
        if (full) {
#pragma unroll
            for (int v = 0; v < 8; ++v) yb[(long long)v * COLS] = acc[v];
        } else {
#pragma unroll
            for (int v = 0; v < 8; ++v) {
                int row = row0 + r0 + (g << 3) + v;
                if (row < N) yb[(long long)v * COLS] = acc[v];
            }
        }
    }
}

// ---------------------------------------------------------------------------
// Edge aggregation: agg[dst] += Y[src] over masked edges. One wave per edge,
// vector gather per lane, fp32 atomics land in L2 (node arrays are L2-resident).
// ---------------------------------------------------------------------------
template <int D>
__global__ __launch_bounds__(256) void
edge_agg_kernel(const int* __restrict__ src, const int* __restrict__ dst,
                const unsigned char* __restrict__ mask,
                const float* __restrict__ Yin, float* __restrict__ agg, int E) {
    int gid = blockIdx.x * blockDim.x + threadIdx.x;
    int e = gid >> 5;
    int lane = gid & 31;
    if (e >= E) return;
    if (!mask[e]) return;
    const long long s = (long long)src[e] * D;
    const long long d = (long long)dst[e] * D;
    if constexpr (D == 128) {
        float4 v = *(const float4*)(Yin + s + lane * 4);
        float* p = agg + d + lane * 4;
        __hip_atomic_fetch_add(p + 0, v.x, __ATOMIC_RELAXED, __HIP_MEMORY_SCOPE_AGENT);
        __hip_atomic_fetch_add(p + 1, v.y, __ATOMIC_RELAXED, __HIP_MEMORY_SCOPE_AGENT);
        __hip_atomic_fetch_add(p + 2, v.z, __ATOMIC_RELAXED, __HIP_MEMORY_SCOPE_AGENT);
        __hip_atomic_fetch_add(p + 3, v.w, __ATOMIC_RELAXED, __HIP_MEMORY_SCOPE_AGENT);
    } else {
        float2 v = *(const float2*)(Yin + s + lane * 2);
        float* p = agg + d + lane * 2;
        __hip_atomic_fetch_add(p + 0, v.x, __ATOMIC_RELAXED, __HIP_MEMORY_SCOPE_AGENT);
        __hip_atomic_fetch_add(p + 1, v.y, __ATOMIC_RELAXED, __HIP_MEMORY_SCOPE_AGENT);
    }
}

// ---------------------------------------------------------------------------
// BatchNorm column stats over h = agg*in_norm + bias.
// ---------------------------------------------------------------------------
template <int D>
__global__ __launch_bounds__(256) void
bn_stats_kernel(const float* __restrict__ agg, const float* __restrict__ in_norm,
                const float* __restrict__ bias, float* __restrict__ sum,
                float* __restrict__ sumsq, int N) {
    constexpr int NR = 256 / D;
    const int tid = threadIdx.x;
    const int col = tid % D;
    const int rg  = tid / D;
    const int rbase = blockIdx.x * 64;
    const float b = bias[col];
    float s = 0.0f, q = 0.0f;
    for (int r = rbase + rg; r < rbase + 64 && r < N; r += NR) {
        float h = agg[(long long)r * D + col] * in_norm[r] + b;
        s += h;
        q += h * h;
    }
    __hip_atomic_fetch_add(&sum[col], s, __ATOMIC_RELAXED, __HIP_MEMORY_SCOPE_AGENT);
    __hip_atomic_fetch_add(&sumsq[col], q, __ATOMIC_RELAXED, __HIP_MEMORY_SCOPE_AGENT);
}

// fold stats + gamma/beta into affine (a, b): bn(h) = h*a + b
__global__ void bn_finalize_kernel(const float* __restrict__ sum,
                                   const float* __restrict__ sumsq,
                                   const float* __restrict__ gamma,
                                   const float* __restrict__ beta,
                                   float* __restrict__ a, float* __restrict__ b,
                                   int D, float invN) {
    int j = blockIdx.x * blockDim.x + threadIdx.x;
    if (j >= D) return;
    float mu = sum[j] * invN;
    float var = sumsq[j] * invN - mu * mu;
    float aj = gamma[j] * __frsqrt_rn(var + BN_EPS);
    a[j] = aj;
    b[j] = beta[j] - mu * aj;
}

// Hs(bf16) = relu(bn(agg*in_norm + b1)) * out_norm  (ready-scaled A for GEMM2;
// only rows < N written, padded tail stays zero from scale_to_bf16_kernel)
__global__ void bn_relu_scale_bf16_kernel(const float* __restrict__ agg,
                                          const float* __restrict__ in_norm,
                                          const float* __restrict__ bias,
                                          const float* __restrict__ a,
                                          const float* __restrict__ b,
                                          const float* __restrict__ out_norm,
                                          unsigned short* __restrict__ Hs, int N) {
    long long i = (long long)blockIdx.x * blockDim.x + threadIdx.x;
    long long total = (long long)N * D_H;
    long long stride = (long long)gridDim.x * blockDim.x;
    for (; i < total; i += stride) {
        int r = (int)(i >> 7);
        int c = (int)(i & (D_H - 1));
        float h = agg[i] * in_norm[r] + bias[c];
        float z = fmaxf(h * a[c] + b[c], 0.0f);
        Hs[i] = f2bf_bits(z * out_norm[r]);
    }
}

// final: bn2 affine + log_softmax over 64 classes. One wave per node,
// 2 classes per lane, wave32 shuffle reductions.
__global__ __launch_bounds__(256) void
bn_logsoftmax_kernel(const float* __restrict__ agg,
                     const float* __restrict__ in_norm,
                     const float* __restrict__ bias,
                     const float* __restrict__ a, const float* __restrict__ b,
                     float* __restrict__ out, int N) {
    int gid = blockIdx.x * blockDim.x + threadIdx.x;
    int node = gid >> 5;
    int lane = gid & 31;
    if (node >= N) return;
    const float inn = in_norm[node];
    const long long base = (long long)node * D_OUT;
    int j0 = lane, j1 = lane + 32;
    float z0 = (agg[base + j0] * inn + bias[j0]) * a[j0] + b[j0];
    float z1 = (agg[base + j1] * inn + bias[j1]) * a[j1] + b[j1];
    float m = fmaxf(z0, z1);
#pragma unroll
    for (int off = 16; off >= 1; off >>= 1)
        m = fmaxf(m, __shfl_xor(m, off, 32));
    float se = __expf(z0 - m) + __expf(z1 - m);
#pragma unroll
    for (int off = 16; off >= 1; off >>= 1)
        se += __shfl_xor(se, off, 32);
    float lse = __logf(se);
    out[base + j0] = z0 - m - lse;
    out[base + j1] = z1 - m - lse;
}

// ---------------------------------------------------------------------------
// Launch
// ---------------------------------------------------------------------------
extern "C" void kernel_launch(void* const* d_in, const int* in_sizes, int n_in,
                              void* d_out, int out_size, void* d_ws, size_t ws_size,
                              hipStream_t stream) {
    const float* features = (const float*)d_in[0];
    const float* W1     = (const float*)d_in[1];
    const float* b1     = (const float*)d_in[2];
    const float* W2     = (const float*)d_in[3];
    const float* b2     = (const float*)d_in[4];
    const float* gamma1 = (const float*)d_in[5];
    const float* beta1  = (const float*)d_in[6];
    const float* gamma2 = (const float*)d_in[7];
    const float* beta2  = (const float*)d_in[8];
    const int* edge_src = (const int*)d_in[9];
    const int* edge_dst = (const int*)d_in[10];
    const unsigned char* edge_mask = (const unsigned char*)d_in[11];

    const int N = in_sizes[0] / D_IN;
    const int E = in_sizes[9];
    const int gemm_blocks = (N + 127) / 128;
    const long long Npad = (long long)gemm_blocks * 128;
    float* outp = (float*)d_out;

    // ---- workspace layout (256B-aligned regions) ----
    char* base = (char*)d_ws;
    size_t off = 0;
    auto take = [&](size_t bytes) {
        char* p = base + off;
        off = (off + bytes + 255) & ~(size_t)255;
        return p;
    };
    float* out_norm = (float*)take(sizeof(float) * N);
    float* in_norm  = (float*)take(sizeof(float) * N);
    float* bufA  = (float*)take(sizeof(float) * (size_t)N * D_H);    // Y1
    float* bufB  = (float*)take(sizeof(float) * (size_t)N * D_H);    // agg1
    float* bufC  = (float*)take(sizeof(float) * (size_t)N * D_OUT);  // Y2
    float* bufD  = (float*)take(sizeof(float) * (size_t)N * D_OUT);  // agg2
    float* stats = (float*)take(sizeof(float) * 768);
    float* sum1 = stats;        float* sq1 = stats + 128;
    float* a1   = stats + 256;  float* s1  = stats + 384;
    float* sum2 = stats + 512;  float* sq2 = stats + 576;
    float* a2   = stats + 640;  float* s2  = stats + 704;
    unsigned short* W1p = (unsigned short*)take(sizeof(short) * D_IN * D_H);
    unsigned short* W2p = (unsigned short*)take(sizeof(short) * D_H * D_OUT);
    unsigned short* Xbf = (unsigned short*)take(sizeof(short) * (size_t)Npad * D_IN);

    const int T = 256;
    const float invN = 1.0f / (float)N;

    // 1) zero: degrees, agg buffers, stats accumulators
    zero_f32_kernel<<<1024, T, 0, stream>>>(out_norm, N);
    zero_f32_kernel<<<1024, T, 0, stream>>>(in_norm, N);
    zero_f32_kernel<<<8192, T, 0, stream>>>(bufB, (long long)N * D_H);
    zero_f32_kernel<<<8192, T, 0, stream>>>(bufD, (long long)N * D_OUT);
    zero_f32_kernel<<<1, T, 0, stream>>>(stats, 768);

    // 2) fragment-packed bf16 weights (L2-resident)
    pack_w_kernel<<<(D_IN * D_H + T - 1) / T, T, 0, stream>>>(W1, W1p, D_H, D_IN * D_H);
    pack_w_kernel<<<(D_H * D_OUT + T - 1) / T, T, 0, stream>>>(W2, W2p, D_OUT, D_H * D_OUT);

    // 3) masked-subgraph degrees -> norms
    degree_kernel<<<(E + T - 1) / T, T, 0, stream>>>(edge_src, edge_dst, edge_mask,
                                                     out_norm, in_norm, E);
    norm_kernel<<<(N + T - 1) / T, T, 0, stream>>>(out_norm, N);
    norm_kernel<<<(N + T - 1) / T, T, 0, stream>>>(in_norm, N);

    const int agg_blocks = (E * 32 + T - 1) / T;
    const int bn_blocks  = (N + 63) / 64;

    // 4) pre-scale X -> bf16 (padded), then GraphConv1 projection [WMMA bf16]
    scale_to_bf16_kernel<<<8192, T, 0, stream>>>(features, out_norm, Xbf, N,
                                                 Npad * D_IN);
    gemm_wmma_kernel<D_H><<<gemm_blocks, T, 0, stream>>>(Xbf, W1p, bufA, N);

    // 5) edge aggregate 1
    edge_agg_kernel<D_H><<<agg_blocks, T, 0, stream>>>(edge_src, edge_dst, edge_mask,
                                                       bufA, bufB, E);

    // 6) BN1 stats + finalize + apply (relu + out_norm pre-scale, bf16 out)
    bn_stats_kernel<D_H><<<bn_blocks, T, 0, stream>>>(bufB, in_norm, b1, sum1, sq1, N);
    bn_finalize_kernel<<<1, 128, 0, stream>>>(sum1, sq1, gamma1, beta1, a1, s1, D_H, invN);
    bn_relu_scale_bf16_kernel<<<8192, T, 0, stream>>>(bufB, in_norm, b1, a1, s1,
                                                      out_norm, Xbf, N);

    // 7) GraphConv2 projection [WMMA bf16]
    gemm_wmma_kernel<D_OUT><<<gemm_blocks, T, 0, stream>>>(Xbf, W2p, bufC, N);

    // 8) edge aggregate 2
    edge_agg_kernel<D_OUT><<<agg_blocks, T, 0, stream>>>(edge_src, edge_dst, edge_mask,
                                                         bufC, bufD, E);

    // 9) BN2 stats + finalize + log_softmax head
    bn_stats_kernel<D_OUT><<<bn_blocks, T, 0, stream>>>(bufD, in_norm, b2, sum2, sq2, N);
    bn_finalize_kernel<<<1, 64, 0, stream>>>(sum2, sq2, gamma2, beta2, a2, s2, D_OUT, invN);
    bn_logsoftmax_kernel<<<(N * 32 + T - 1) / T, T, 0, stream>>>(bufD, in_norm, b2,
                                                                 a2, s2, outp, N);
}
